// RESEGNN_40638980555249
// MI455X (gfx1250) — matrix operand order repaired and checked by reading, not compiled, not run on hardware
//
#include <hip/hip_runtime.h>
#include <hip/hip_bf16.h>
#include <stdint.h>

// ---------------------------------------------------------------------------
// EGNN forward for MI455X (gfx1250): fused bf16-WMMA edge/node MLPs.
//   N_NODES=20000, N_EDGES=640000, HID=128, 4 layers.
//   - batched B-tile LDS loads -> back-to-back v_wmma chains
//   - A-tile double buffering (global gathers)
//   - async weight staging via global_load_async_to_lds_b128 (ASYNCcnt)
// ---------------------------------------------------------------------------

#define HID 128

typedef __bf16 bf16;
typedef __attribute__((ext_vector_type(16))) __bf16 v16bf;
typedef __attribute__((ext_vector_type(8)))  __bf16 v8bf;
typedef __attribute__((ext_vector_type(8)))  float  v8f;

__device__ __forceinline__ v8f wmma_bf16(v16bf a, v16bf b, v8f c) {
  // D = A(16x32) * B(32x16) + C, fp32 accumulate
  return __builtin_amdgcn_wmma_f32_16x16x32_bf16(false, a, false, b, (short)0, c,
                                                 false, false);
}

__device__ __forceinline__ float silu_f(float v) {
  return v / (1.0f + __expf(-v));
}

__device__ __forceinline__ v16bf combine8(v8bf lo, v8bf hi) {
  v16bf a;
#pragma unroll
  for (int i = 0; i < 8; ++i) { a[i] = lo[i]; a[i + 8] = hi[i]; }
  return a;
}

// A-tile (16x32, 16-bit): lane<16 row M=lane, halves 0..7 -> K=kbase+0..7,
// halves 8..15 -> K=kbase+16..23; lane>=16 same row group with +8 K offset.
__device__ __forceinline__ v16bf load_a_bf16(const bf16* rowPtr, int kbase, int lane) {
  const int k0 = kbase + ((lane & 16) ? 8 : 0);
  v8bf lo = *(const v8bf*)(rowPtr + k0);
  v8bf hi = *(const v8bf*)(rowPtr + k0 + 16);
  return combine8(lo, hi);
}

__device__ __forceinline__ v16bf load_a_f32(const float* rowPtr, int kbase, int lane) {
  const int k0 = kbase + ((lane & 16) ? 8 : 0);
  v16bf a;
#pragma unroll
  for (int i = 0; i < 8; ++i) a[i] = (bf16)rowPtr[k0 + i];
#pragma unroll
  for (int i = 0; i < 8; ++i) a[i + 8] = (bf16)rowPtr[k0 + 16 + i];
  return a;
}

// B-tile (32x16, 16-bit) from W^T stored [N][Ks] (K contiguous):
// lanes 0..15 hold K=kbase+0..7,16..23 of column N=nbase+lane,
// lanes 16..31 hold K=kbase+8..15,24..31.
__device__ __forceinline__ v16bf load_b_t(const bf16* wT, int ks, int kbase,
                                          int nbase, int lane) {
  const int n  = nbase + (lane & 15);
  const int k0 = kbase + ((lane & 16) ? 8 : 0);
  const bf16* p = wT + n * ks + k0;
  v8bf lo = *(const v8bf*)p;
  v8bf hi = *(const v8bf*)(p + 16);
  return combine8(lo, hi);
}

__device__ __forceinline__ void wave_lds_fence() {
  // Intra-wave cross-lane visibility through LDS: DS ops are in-order per
  // wave; drain DScnt and stop compiler reordering.
  asm volatile("s_wait_dscnt 0" ::: "memory");
}

// Async DMA: 16 bytes per lane, global -> LDS, tracked by ASYNCcnt.
// Low 32 bits of a generic pointer into LDS are the LDS logical address
// (flat-aperture mapping), as the instruction's VDST LDS-address operand expects.
__device__ __forceinline__ void async_copy16(void* ldsDst, const void* gsrc) {
  const uint32_t loff = (uint32_t)(uintptr_t)ldsDst;
  asm volatile("global_load_async_to_lds_b128 %0, %1, off"
               : : "v"(loff), "v"(gsrc) : "memory");
}
__device__ __forceinline__ void wait_async0() {
  asm volatile("s_wait_asynccnt 0" ::: "memory");
}

// ---------------------------------------------------------------------------
// Fused edge kernel: per 16-edge tile (one wave):
//   radial -> edge1(257->128) -> SiLU -> edge2(128->128) -> SiLU = m
//   m -> atomicAdd magg[row]; m -> coord1 -> SiLU -> dot(coord2) -> scatter x
// ---------------------------------------------------------------------------
__global__ __launch_bounds__(256, 1) void egnn_edge_kernel(
    const float* __restrict__ x, const bf16* __restrict__ hb,
    const int* __restrict__ erow, const int* __restrict__ ecol,
    const bf16* __restrict__ e1T,   // [128][256] bf16 (rows 0..255 of edge1.w, transposed)
    const bf16* __restrict__ e2T,   // [128][128]
    const bf16* __restrict__ c1T,   // [128][128]  (streamed from global, L2/WGP$-hot)
    const float* __restrict__ e1wr, // radial row (row 256) of edge1.w, 128 f32
    const float* __restrict__ e1b, const float* __restrict__ e2b,
    const float* __restrict__ c1b, const float* __restrict__ c2w,
    float* __restrict__ magg, float* __restrict__ xagg, int nTiles) {
  __shared__ __align__(16) bf16 sW1[128 * 256];      // 64 KB
  __shared__ __align__(16) bf16 sW2[128 * 128];      // 32 KB
  __shared__ __align__(16) bf16 sAct[8][16 * HID];   // 32 KB (per-wave 16x128 staging)
  __shared__ float sB1[128], sB2[128], sB3[128], sWr[128], sC2[128];
  __shared__ int   sRow[8][16], sCol[8][16];
  __shared__ float sRad[8][16], sDx[8][16], sDy[8][16], sDz[8][16];

  const int tid = threadIdx.x;
  {  // async-DMA weight staging into LDS (ASYNCcnt path)
    for (int i = tid; i < (128 * 256 * 2) / 16; i += 256)
      async_copy16((char*)sW1 + i * 16, (const char*)e1T + i * 16);
    for (int i = tid; i < (128 * 128 * 2) / 16; i += 256)
      async_copy16((char*)sW2 + i * 16, (const char*)e2T + i * 16);
    if (tid < 128) {
      sB1[tid] = e1b[tid]; sB2[tid] = e2b[tid]; sB3[tid] = c1b[tid];
      sWr[tid] = e1wr[tid]; sC2[tid] = c2w[tid];
    }
    // pull the streamed coord1^T matrix toward WGP caches
    for (int i = tid * 128; i < 128 * 128 * 2; i += 256 * 128)
      __builtin_prefetch((const char*)c1T + i, 0, 0);
    wait_async0();
  }
  __syncthreads();

  const int wave = tid >> 5;
  const int lane = tid & 31;
  const int fb   = lane & 15;
  const int hi8  = (lane & 16) ? 8 : 0;
  bf16* act = &sAct[wave][0];

  for (int tile = blockIdx.x * 8 + wave; tile < nTiles; tile += gridDim.x * 8) {
    const int e0 = tile * 16;
    if (lane < 16) {  // per-edge geometry for this tile
      const int e = e0 + lane;
      const int r = erow[e], c = ecol[e];
      const float dx = x[3 * r + 0] - x[3 * c + 0];
      const float dy = x[3 * r + 1] - x[3 * c + 1];
      const float dz = x[3 * r + 2] - x[3 * c + 2];
      sRow[wave][lane] = r; sCol[wave][lane] = c;
      sDx[wave][lane] = dx; sDy[wave][lane] = dy; sDz[wave][lane] = dz;
      sRad[wave][lane] = dx * dx + dy * dy + dz * dz;
    }
    wave_lds_fence();

    // ---- edge1: [16 x 257] @ [257 x 128]; radial column folded in epilogue.
    // A double-buffered (global gathers), B batched (one LDS clause -> 8 WMMA).
    v8f acc[8] = {};
    {
      v16bf a = load_a_bf16(hb + (size_t)sRow[wave][fb] * HID, 0, lane);
#pragma unroll
      for (int kt = 0; kt < 8; ++kt) {
        v16bf b[8];
#pragma unroll
        for (int nt = 0; nt < 8; ++nt)
          b[nt] = load_b_t(sW1, 256, kt * 32, nt * 16, lane);
        v16bf aNext = a;
        if (kt < 7) {
          const int node = (kt + 1 < 4) ? sRow[wave][fb] : sCol[wave][fb];
          aNext = load_a_bf16(hb + (size_t)node * HID, ((kt + 1) & 3) * 32, lane);
        }
#pragma unroll
        for (int nt = 0; nt < 8; ++nt) acc[nt] = wmma_bf16(a, b[nt], acc[nt]);
        a = aNext;
      }
    }
#pragma unroll
    for (int nt = 0; nt < 8; ++nt) {
      const int f = nt * 16 + fb;
      const float b = sB1[f], wr = sWr[f];
#pragma unroll
      for (int v = 0; v < 8; ++v) {
        const int m = v + hi8;
        act[m * HID + f] = (bf16)silu_f(acc[nt][v] + b + sRad[wave][m] * wr);
      }
    }
    wave_lds_fence();

    // ---- edge2: [16x128] @ [128x128] -> m
    v8f acc2[8] = {};
#pragma unroll
    for (int kt = 0; kt < 4; ++kt) {
      const v16bf a = load_a_bf16(act + fb * HID, kt * 32, lane);
      v16bf b[8];
#pragma unroll
      for (int nt = 0; nt < 8; ++nt)
        b[nt] = load_b_t(sW2, 128, kt * 32, nt * 16, lane);
#pragma unroll
      for (int nt = 0; nt < 8; ++nt) acc2[nt] = wmma_bf16(a, b[nt], acc2[nt]);
    }
    wave_lds_fence();
#pragma unroll
    for (int nt = 0; nt < 8; ++nt) {
      const int f = nt * 16 + fb;
      const float b = sB2[f];
#pragma unroll
      for (int v = 0; v < 8; ++v) {
        const int m = v + hi8;
        const float mv = silu_f(acc2[nt][v] + b);
        atomicAdd(&magg[(size_t)sRow[wave][m] * HID + f], mv);  // L2-resident
        act[m * HID + f] = (bf16)mv;
      }
    }
    wave_lds_fence();

    // ---- coord1 (+ fused coord2 dot); B streamed from global (cache-hot)
    v8f acc3[8] = {};
#pragma unroll
    for (int kt = 0; kt < 4; ++kt) {
      const v16bf a = load_a_bf16(act + fb * HID, kt * 32, lane);
      v16bf b[8];
#pragma unroll
      for (int nt = 0; nt < 8; ++nt)
        b[nt] = load_b_t(c1T, 128, kt * 32, nt * 16, lane);
#pragma unroll
      for (int nt = 0; nt < 8; ++nt) acc3[nt] = wmma_bf16(a, b[nt], acc3[nt]);
    }
    float part[8] = {0, 0, 0, 0, 0, 0, 0, 0};
#pragma unroll
    for (int nt = 0; nt < 8; ++nt) {
      const int f = nt * 16 + fb;
      const float b = sB3[f], w = sC2[f];
#pragma unroll
      for (int v = 0; v < 8; ++v) part[v] += silu_f(acc3[nt][v] + b) * w;
    }
#pragma unroll
    for (int off = 1; off < 16; off <<= 1)
#pragma unroll
      for (int v = 0; v < 8; ++v) part[v] += __shfl_xor(part[v], off, 32);
    if (fb == 0) {  // lane0 -> edges 0..7, lane16 -> edges 8..15
#pragma unroll
      for (int v = 0; v < 8; ++v) {
        const int m = v + hi8;
        const float s = part[v];
        const int r = sRow[wave][m];
        atomicAdd(&xagg[3 * r + 0], sDx[wave][m] * s);
        atomicAdd(&xagg[3 * r + 1], sDy[wave][m] * s);
        atomicAdd(&xagg[3 * r + 2], sDz[wave][m] * s);
      }
    }
  }
}

// ---------------------------------------------------------------------------
// Fused node kernel: [h || magg](256) -> node1 -> SiLU -> node2, residual h+=
// ---------------------------------------------------------------------------
__global__ __launch_bounds__(256, 1) void egnn_node_kernel(
    const float* __restrict__ magg,
    const bf16* __restrict__ n1T,  // [128][256]
    const bf16* __restrict__ n2T,  // [128][128]
    const float* __restrict__ n1b, const float* __restrict__ n2b,
    float* __restrict__ h, bf16* __restrict__ hb, int nTiles) {
  __shared__ __align__(16) bf16 sW1[128 * 256];
  __shared__ __align__(16) bf16 sW2[128 * 128];
  __shared__ __align__(16) bf16 sAct[8][16 * HID];
  __shared__ float sB1[128], sB2[128];
  const int tid = threadIdx.x;
  {
    for (int i = tid; i < (128 * 256 * 2) / 16; i += 256)
      async_copy16((char*)sW1 + i * 16, (const char*)n1T + i * 16);
    for (int i = tid; i < (128 * 128 * 2) / 16; i += 256)
      async_copy16((char*)sW2 + i * 16, (const char*)n2T + i * 16);
    if (tid < 128) { sB1[tid] = n1b[tid]; sB2[tid] = n2b[tid]; }
    wait_async0();
  }
  __syncthreads();
  const int wave = tid >> 5, lane = tid & 31;
  const int fb = lane & 15, hi8 = (lane & 16) ? 8 : 0;
  bf16* act = &sAct[wave][0];

  for (int tile = blockIdx.x * 8 + wave; tile < nTiles; tile += gridDim.x * 8) {
    const int n0 = tile * 16;
    const int nodeRow = n0 + fb;
    v8f acc[8] = {};
    {
      v16bf a = load_a_f32(h + (size_t)nodeRow * HID, 0, lane);
#pragma unroll
      for (int kt = 0; kt < 8; ++kt) {
        v16bf b[8];
#pragma unroll
        for (int nt = 0; nt < 8; ++nt)
          b[nt] = load_b_t(sW1, 256, kt * 32, nt * 16, lane);
        v16bf aNext = a;
        if (kt < 7) {
          const float* src = (kt + 1 < 4) ? (h + (size_t)nodeRow * HID)
                                          : (magg + (size_t)nodeRow * HID);
          aNext = load_a_f32(src, ((kt + 1) & 3) * 32, lane);
        }
#pragma unroll
        for (int nt = 0; nt < 8; ++nt) acc[nt] = wmma_bf16(a, b[nt], acc[nt]);
        a = aNext;
      }
    }
#pragma unroll
    for (int nt = 0; nt < 8; ++nt) {
      const int f = nt * 16 + fb;
      const float b = sB1[f];
#pragma unroll
      for (int v = 0; v < 8; ++v) {
        const int m = v + hi8;
        act[m * HID + f] = (bf16)silu_f(acc[nt][v] + b);
      }
    }
    wave_lds_fence();
    v8f acc2[8] = {};
#pragma unroll
    for (int kt = 0; kt < 4; ++kt) {
      const v16bf a = load_a_bf16(act + fb * HID, kt * 32, lane);
      v16bf b[8];
#pragma unroll
      for (int nt = 0; nt < 8; ++nt)
        b[nt] = load_b_t(sW2, 128, kt * 32, nt * 16, lane);
#pragma unroll
      for (int nt = 0; nt < 8; ++nt) acc2[nt] = wmma_bf16(a, b[nt], acc2[nt]);
    }
#pragma unroll
    for (int nt = 0; nt < 8; ++nt) {
      const int f = nt * 16 + fb;
      const float b = sB2[f];
#pragma unroll
      for (int v = 0; v < 8; ++v) {
        const int m = v + hi8;
        const size_t idx = (size_t)(n0 + m) * HID + f;
        const float val = h[idx] + acc2[nt][v] + b;  // residual
        h[idx] = val;
        hb[idx] = (bf16)val;
      }
    }
  }
}

// ----------------------------- small helpers -------------------------------
__global__ void zero_kernel(float* __restrict__ p, int n) {
  int i = blockIdx.x * blockDim.x + threadIdx.x;
  if (i < n) p[i] = 0.0f;
}

__global__ void count_kernel(const int* __restrict__ erow, float* __restrict__ cnt, int e) {
  int i = blockIdx.x * blockDim.x + threadIdx.x;
  if (i < e) atomicAdd(&cnt[erow[i]], 1.0f);
}

__global__ void embed_kernel(const float* __restrict__ nf, const float* __restrict__ w,
                             const float* __restrict__ b, float* __restrict__ h,
                             bf16* __restrict__ hb, int n) {
  int idx = blockIdx.x * blockDim.x + threadIdx.x;
  if (idx >= n * HID) return;
  const int node = idx >> 7, f = idx & 127;
  float s = b[f];
  const float* row = nf + node * 21;
#pragma unroll
  for (int k = 0; k < 21; ++k) s += row[k] * w[k * HID + f];
  h[idx] = s;
  hb[idx] = (bf16)s;
}

// W row-major [K][N] (f32) -> W^T [N][K] (bf16)
__global__ void wT_kernel(const float* __restrict__ w, bf16* __restrict__ wT,
                          int K, int N) {
  int i = blockIdx.x * blockDim.x + threadIdx.x;
  if (i >= K * N) return;
  const int n = i / K, k = i - n * K;
  wT[i] = (bf16)w[k * N + n];
}

__global__ void coord_update_kernel(float* __restrict__ x, const float* __restrict__ xagg,
                                    const float* __restrict__ cnt, int n3) {
  int i = blockIdx.x * blockDim.x + threadIdx.x;
  if (i < n3) x[i] += xagg[i] / fmaxf(cnt[i / 3], 1.0f);
}

// Final head: only node 0 of h matters after emb_out.
__global__ void head_kernel(const float* __restrict__ h, const float* __restrict__ eow,
                            const float* __restrict__ eob, const float* __restrict__ h1w,
                            const float* __restrict__ h1b, const float* __restrict__ h2w,
                            const float* __restrict__ h2b, float* __restrict__ out) {
  __shared__ float h0[128], c[128], z[128];
  const int t = threadIdx.x;  // 128 threads
  h0[t] = h[t];
  __syncthreads();
  float s = eob[t];
  for (int k = 0; k < 128; ++k) s += h0[k] * eow[k * 128 + t];
  c[t] = s;
  __syncthreads();
  s = h1b[t];
  for (int k = 0; k < 128; ++k) s += c[k] * h1w[k * 128 + t];
  z[t] = fmaxf(s, 0.0f);
  __syncthreads();
  if (t < 20) {
    s = h2b[t];
    for (int k = 0; k < 128; ++k) s += z[k] * h2w[k * 20 + t];
    out[t] = s;
  }
}

// ---------------------------------------------------------------------------
extern "C" void kernel_launch(void* const* d_in, const int* in_sizes, int n_in,
                              void* d_out, int out_size, void* d_ws, size_t ws_size,
                              hipStream_t stream) {
  (void)n_in; (void)out_size; (void)ws_size;
  const float* node_feats = (const float*)d_in[0];
  const float* pos        = (const float*)d_in[1];
  const int*   eidx       = (const int*)d_in[2];
  const int N = in_sizes[1] / 3;
  const int E = in_sizes[2] / 2;
  const int* erow = eidx;
  const int* ecol = eidx + E;
  const float* emb_in_w = (const float*)d_in[3];
  const float* emb_in_b = (const float*)d_in[4];
  // Per-layer params (dict insertion order: edge1,edge2,node1,node2,coord1,coord2)
  struct Layer {
    const float *e1w, *e1b, *e2w, *e2b, *n1w, *n1b, *n2w, *n2b, *c1w, *c1b, *c2w;
  } L[4];
  for (int l = 0; l < 4; ++l) {
    const int b = 5 + l * 11;
    L[l].e1w = (const float*)d_in[b + 0];  L[l].e1b = (const float*)d_in[b + 1];
    L[l].e2w = (const float*)d_in[b + 2];  L[l].e2b = (const float*)d_in[b + 3];
    L[l].n1w = (const float*)d_in[b + 4];  L[l].n1b = (const float*)d_in[b + 5];
    L[l].n2w = (const float*)d_in[b + 6];  L[l].n2b = (const float*)d_in[b + 7];
    L[l].c1w = (const float*)d_in[b + 8];  L[l].c1b = (const float*)d_in[b + 9];
    L[l].c2w = (const float*)d_in[b + 10];
  }
  const float* emb_out_w = (const float*)d_in[49];
  const float* emb_out_b = (const float*)d_in[50];
  const float* head1_w   = (const float*)d_in[51];
  const float* head1_b   = (const float*)d_in[52];
  const float* head2_w   = (const float*)d_in[53];
  const float* head2_b   = (const float*)d_in[54];

  // ---- workspace carve (all L2-resident working state)
  uintptr_t p = (uintptr_t)d_ws;
  auto carve = [&](size_t bytes) {
    void* r = (void*)p;
    p += (bytes + 255) & ~(size_t)255;
    return r;
  };
  float* h    = (float*)carve((size_t)N * HID * 4);
  float* magg = (float*)carve((size_t)N * HID * 4);
  float* x    = (float*)carve((size_t)N * 3 * 4);
  float* xagg = (float*)carve((size_t)N * 3 * 4);
  float* cnt  = (float*)carve((size_t)N * 4);
  bf16*  hb   = (bf16*)carve((size_t)N * HID * 2);
  bf16 *e1T[4], *e2T[4], *c1T[4], *n1T[4], *n2T[4];
  for (int l = 0; l < 4; ++l) {
    e1T[l] = (bf16*)carve(128 * 256 * 2);
    e2T[l] = (bf16*)carve(128 * 128 * 2);
    c1T[l] = (bf16*)carve(128 * 128 * 2);
    n1T[l] = (bf16*)carve(128 * 256 * 2);
    n2T[l] = (bf16*)carve(128 * 128 * 2);
  }

  // ---- preprocessing
  zero_kernel<<<(N + 255) / 256, 256, 0, stream>>>(cnt, N);
  count_kernel<<<(E + 255) / 256, 256, 0, stream>>>(erow, cnt, E);
  hipMemcpyAsync(x, pos, (size_t)N * 3 * 4, hipMemcpyDeviceToDevice, stream);
  embed_kernel<<<(N * HID + 255) / 256, 256, 0, stream>>>(node_feats, emb_in_w,
                                                          emb_in_b, h, hb, N);
  for (int l = 0; l < 4; ++l) {
    wT_kernel<<<(128 * 256 + 255) / 256, 256, 0, stream>>>(L[l].e1w, e1T[l], 256, 128);
    wT_kernel<<<(128 * 128 + 255) / 256, 256, 0, stream>>>(L[l].e2w, e2T[l], 128, 128);
    wT_kernel<<<(128 * 128 + 255) / 256, 256, 0, stream>>>(L[l].c1w, c1T[l], 128, 128);
    wT_kernel<<<(128 * 256 + 255) / 256, 256, 0, stream>>>(L[l].n1w, n1T[l], 256, 128);
    wT_kernel<<<(128 * 128 + 255) / 256, 256, 0, stream>>>(L[l].n2w, n2T[l], 128, 128);
  }

  const int eTiles = E / 16;  // 40000 (E divisible by 16)
  const int nTiles = N / 16;  // 1250
  const int gridE = 1024, gridN = 160;

  for (int l = 0; l < 4; ++l) {
    zero_kernel<<<(N * HID + 255) / 256, 256, 0, stream>>>(magg, N * HID);
    zero_kernel<<<(N * 3 + 255) / 256, 256, 0, stream>>>(xagg, N * 3);
    egnn_edge_kernel<<<gridE, 256, 0, stream>>>(
        x, hb, erow, ecol, e1T[l], e2T[l], c1T[l],
        L[l].e1w + 256 * 128,  // radial row of edge1.w
        L[l].e1b, L[l].e2b, L[l].c1b, L[l].c2w, magg, xagg, eTiles);
    coord_update_kernel<<<(N * 3 + 255) / 256, 256, 0, stream>>>(x, xagg, cnt, N * 3);
    egnn_node_kernel<<<gridN, 256, 0, stream>>>(magg, n1T[l], n2T[l], L[l].n1b,
                                                L[l].n2b, h, hb, nTiles);
  }
  head_kernel<<<1, 128, 0, stream>>>(h, emb_out_w, emb_out_b, head1_w, head1_b,
                                     head2_w, head2_b, (float*)d_out);
}
